// ConditionalVectorQuantizedVAE_51196010169094
// MI455X (gfx1250) — compile-verified, matrix operand order checked by reading, not run on hardware
//
#include <hip/hip_runtime.h>
#include <cstdint>
#include <cstring>

// ---------------------------------------------------------------------------
// BF16 WMMA implicit-GEMM implementation of the conditional VQ-VAE forward.
// All large matmuls (enc conv2, 4 resblocks, VQ scores, dec convT1) run on
// v_wmma_f32_16x16x32_bf16 with fp32 accumulation.
// Wave tile 64x64 (16 accumulators), tap-hoisted addressing (no divisions in
// the K inner loop), global_prefetch on the weight stream.
// ---------------------------------------------------------------------------

typedef __attribute__((ext_vector_type(16))) __bf16 bf16x16;
typedef __attribute__((ext_vector_type(8)))  float  f32x8;

__device__ __forceinline__ __bf16 f2bf(float f) {
  unsigned u = __float_as_uint(f);
  unsigned r = (u + 0x7FFFu + ((u >> 16) & 1u)) >> 16;
  unsigned short s = (unsigned short)r;
  return __builtin_bit_cast(__bf16, s);
}
__device__ __forceinline__ float bf2f(__bf16 h) {
  unsigned short s = __builtin_bit_cast(unsigned short, h);
  return __uint_as_float(((unsigned)s) << 16);
}

union AFrag { bf16x16 v; float4 q[2]; };

struct Geom {
  int IH, IW, IC;     // input image
  int OH, OW, S;      // output (per-class) spatial, stride
  int ntaps, Ktot, Nout;
  int dy[16], dx[16]; // tap offsets (pad folded in)
  int OYS, OYO, OXS, OXO, OHO, OWO; // output pixel remap (for convT classes)
  int store_bf16;
};

struct PackW { int IC, KH, KW, Nout, Ktot; };

// ---------------------------------------------------------------------------
// Core WMMA implicit-GEMM: D[m, n] = sum_k A[m,k] * B[k,n]
//   A: gathered from NHWC bf16 activations (k = tap*IC + channel)
//   B: pre-swizzled [kc][ntile][lane][16] bf16
// Block: 256 threads = 8 waves in 2(M) x 4(N) grid; wave tile = 64x64;
// block tile = 128(M) x 256(N).
// ---------------------------------------------------------------------------
__global__ __launch_bounds__(256) void wmma_gemm(
    const __bf16* __restrict__ in, const __bf16* __restrict__ Bp,
    float* __restrict__ outF, __bf16* __restrict__ outH,
    const int* __restrict__ cond, int tab_stride, Geom g)
{
  const int lane  = threadIdx.x & 31;
  const int wave  = threadIdx.x >> 5;
  const int waveM = wave >> 2;        // 0..1
  const int waveN = wave & 3;         // 0..3
  const int m0 = blockIdx.x * 128 + waveM * 64;
  const int n0 = blockIdx.y * 256 + waveN * 64;
  const int lr = lane & 15, khalf = lane >> 4;
  const int OHW = g.OH * g.OW;

  int nimg[4], ohh[4], ow0[4];
#pragma unroll
  for (int i = 0; i < 4; ++i) {
    int m = m0 + i * 16;
    nimg[i] = m / OHW;
    int rem = m - nimg[i] * OHW;
    ohh[i] = rem / g.OW;
    ow0[i] = rem - ohh[i] * g.OW;
  }

  const __bf16* bptr = Bp;
  if (cond) bptr += (size_t)cond[nimg[0]] * tab_stride;

  const int cpt = g.IC >> 5;          // 32-wide chunks per tap (== 8 here)
  const int ntile0 = n0 >> 4;
  const int ntilesTot = g.Nout >> 4;
  const size_t bChunkStride = (size_t)ntilesTot * 512;  // elems per k-chunk row

  f32x8 acc[4][4] = {};

  for (int tap = 0; tap < g.ntaps; ++tap) {
    // Per-tap A row pointers + validity (padding); uniform within the tap.
    const __bf16* arow[4];
    bool av[4];
#pragma unroll
    for (int i = 0; i < 4; ++i) {
      int ih = ohh[i] * g.S + g.dy[tap];
      int iw = (ow0[i] + lr) * g.S + g.dx[tap];
      av[i] = (ih >= 0) & (ih < g.IH) & (iw >= 0) & (iw < g.IW);
      arow[i] = in + (((size_t)nimg[i] * g.IH + ih) * g.IW + iw) * g.IC + khalf * 8;
    }
    const __bf16* brow = bptr + (size_t)(tap * cpt) * bChunkStride
                              + (size_t)ntile0 * 512 + lane * 16;
    if (tap + 1 < g.ntaps)   // stream next tap's weights behind current WMMAs
      __builtin_prefetch((const void*)(bptr + (size_t)((tap + 1) * cpt) * bChunkStride
                                       + (size_t)ntile0 * 512 + lane * 16), 0, 1);

    for (int cc = 0; cc < cpt; ++cc) {
      AFrag a[4];
#pragma unroll
      for (int i = 0; i < 4; ++i) {
        float4 z = {0.f, 0.f, 0.f, 0.f};
        a[i].q[0] = z; a[i].q[1] = z;
        if (av[i]) {
          const float4* p = reinterpret_cast<const float4*>(arow[i] + cc * 32);
          a[i].q[0] = p[0];
          a[i].q[1] = p[2];   // +16 channels
        }
      }
      AFrag b[4];
#pragma unroll
      for (int j = 0; j < 4; ++j) {
        const float4* p = reinterpret_cast<const float4*>(brow + j * 512);
        b[j].q[0] = p[0];
        b[j].q[1] = p[1];
      }
      brow += bChunkStride;

#pragma unroll
      for (int i = 0; i < 4; ++i)
#pragma unroll
        for (int j = 0; j < 4; ++j)
          acc[i][j] = __builtin_amdgcn_wmma_f32_16x16x32_bf16(
              false, a[i].v, false, b[j].v, (short)0, acc[i][j], false, false);
    }
  }

  // Store: D layout lanes 0-15 -> N=lane, M=r ; lanes 16-31 -> N=lane-16, M=r+8
#pragma unroll
  for (int i = 0; i < 4; ++i) {
    int oy = ohh[i] * g.OYS + g.OYO;
#pragma unroll
    for (int r = 0; r < 8; ++r) {
      int row = r + khalf * 8;
      int ox = (ow0[i] + row) * g.OXS + g.OXO;
      size_t off = (((size_t)nimg[i] * g.OHO + oy) * g.OWO + ox) * (size_t)g.Nout;
#pragma unroll
      for (int j = 0; j < 4; ++j) {
        int ch = n0 + j * 16 + lr;
        float v = acc[i][j][r];
        if (g.store_bf16) outH[off + ch] = f2bf(v);
        else              outF[off + ch] = v;
      }
    }
  }
}

// ---------------------------------------------------------------------------
// Weight / codebook packing into WMMA B-fragment layout [kc][ntile][lane][16]
// ---------------------------------------------------------------------------
__global__ __launch_bounds__(256) void pack_conv_w(
    const float* __restrict__ w, __bf16* __restrict__ Bp, PackW p, int total)
{
  int idx = blockIdx.x * 256 + threadIdx.x;
  if (idx >= total) return;
  int e = idx & 15;
  int lane = (idx >> 4) & 31;
  int t = idx >> 9;
  int ntiles = p.Nout >> 4;
  int nt = t % ntiles, kc = t / ntiles;
  int lr = lane & 15, khalf = lane >> 4;
  int k = kc * 32 + ((e >> 3) << 4) + khalf * 8 + (e & 7);
  int n = nt * 16 + lr;
  float v = 0.f;
  if (k < p.Ktot) {
    int tap = k / p.IC, c = k - tap * p.IC;
    int kh = tap / p.KW, kw = tap - kh * p.KW;
    v = w[(((size_t)n * p.IC + c) * p.KH + kh) * p.KW + kw];
  }
  Bp[idx] = f2bf(v);
}

// convT weight pack: w is [Cin=256][Cout][4][4]; class (a,bx): tap t=tky*2+tkx,
// ky=a+2*tky, kx=bx+2*tkx; value = w[c][n][3-ky][3-kx]
__global__ __launch_bounds__(256) void pack_convT(
    const float* __restrict__ w, __bf16* __restrict__ Bp,
    int Cout, int a, int bx, int total)
{
  int idx = blockIdx.x * 256 + threadIdx.x;
  if (idx >= total) return;
  int e = idx & 15;
  int lane = (idx >> 4) & 31;
  int t = idx >> 9;
  int ntiles = Cout >> 4;
  int nt = t % ntiles, kc = t / ntiles;
  int lr = lane & 15, khalf = lane >> 4;
  int k = kc * 32 + ((e >> 3) << 4) + khalf * 8 + (e & 7);
  int n = nt * 16 + lr;
  int tap = k >> 8, c = k & 255;
  int tky = tap >> 1, tkx = tap & 1;
  int ky = a + 2 * tky, kx = bx + 2 * tkx;
  float v = w[(((size_t)c * Cout + n) * 4 + (3 - ky)) * 4 + (3 - kx)];
  Bp[idx] = f2bf(v);
}

// codebook pack: per cond table, B[k=d][n=code] = cb[cond][code][d]
__global__ __launch_bounds__(256) void pack_cb(
    const float* __restrict__ cb, __bf16* __restrict__ Bp)
{
  int idx = blockIdx.x * 256 + threadIdx.x;   // < 16*131072
  int cnd = idx >> 17;
  int rem = idx & 131071;
  int e = rem & 15;
  int lane = (rem >> 4) & 31;
  int t = rem >> 9;
  int nt = t & 31, kc = t >> 5;
  int lr = lane & 15, khalf = lane >> 4;
  int k = kc * 32 + ((e >> 3) << 4) + khalf * 8 + (e & 7);
  int n = nt * 16 + lr;
  Bp[idx] = f2bf(cb[((size_t)cnd * 512 + n) * 256 + k]);
}

// ---------------------------------------------------------------------------
// Elementwise / reduction kernels
// ---------------------------------------------------------------------------
__global__ __launch_bounds__(256) void k_bn_stats(
    const float* __restrict__ xF, const __bf16* __restrict__ xH,
    float* __restrict__ st, int ppb)
{
  int c = threadIdx.x;
  size_t p0 = (size_t)blockIdx.x * ppb;
  float s = 0.f, ss = 0.f;
  for (int i = 0; i < ppb; ++i) {
    size_t off = (p0 + i) * 256 + c;
    float v = xF ? xF[off] : bf2f(xH[off]);
    s += v; ss += v * v;
  }
  atomicAdd(&st[c], s);
  atomicAdd(&st[256 + c], ss);
}

__global__ __launch_bounds__(256) void k_bn_apply(
    const float* __restrict__ xF, const __bf16* __restrict__ xH,
    const float* __restrict__ st, float inv_cnt,
    const float* __restrict__ gam, const float* __restrict__ bet, int do_relu,
    const float* __restrict__ resid, float* __restrict__ outF,
    __bf16* __restrict__ outH, size_t n)
{
  size_t i = (size_t)blockIdx.x * 256 + threadIdx.x;
  if (i >= n) return;
  int c = (int)(i & 255);
  float x = xF ? xF[i] : bf2f(xH[i]);
  float m = st[c] * inv_cnt;
  float var = st[256 + c] * inv_cnt - m * m;
  float y = gam[c] * (x - m) * rsqrtf(var + 1e-5f) + bet[c];
  if (do_relu) y = fmaxf(y, 0.f);
  if (resid) y += resid[i];
  if (outF) outF[i] = y;
  if (outH) outH[i] = f2bf(y);
}

__global__ __launch_bounds__(256) void k_relu_bf16(
    const float* __restrict__ x, __bf16* __restrict__ o, size_t n)
{
  size_t i = (size_t)blockIdx.x * 256 + threadIdx.x;
  if (i < n) o[i] = f2bf(fmaxf(x[i], 0.f));
}

__global__ __launch_bounds__(256) void k_add_bias(
    float* __restrict__ x, const float* __restrict__ b, size_t n)
{
  size_t i = (size_t)blockIdx.x * 256 + threadIdx.x;
  if (i < n) x[i] += b[i & 255];
}

// z_e export: NHWC fp32 -> NCHW fp32 (d_out) + bf16 NHWC (for VQ GEMM)
__global__ __launch_bounds__(256) void k_export_small(
    const float* __restrict__ src, float* __restrict__ dst_nchw,
    __bf16* __restrict__ dstH, size_t n)
{
  size_t i = (size_t)blockIdx.x * 256 + threadIdx.x;
  if (i >= n) return;
  int c = (int)(i & 255);
  size_t p = i >> 8;
  int b = (int)(p >> 10);
  int rem = (int)(p & 1023);
  int y = rem >> 5, x = rem & 31;
  float v = src[i];
  dst_nchw[(((size_t)b * 256 + c) * 32 + y) * 32 + x] = v;
  if (dstH) dstH[i] = f2bf(v);
}

__global__ __launch_bounds__(256) void k_cb_norms(
    const float* __restrict__ cb, float* __restrict__ norms)
{
  int i = blockIdx.x * 256 + threadIdx.x;   // < 8192
  const float* r = cb + (size_t)i * 256;
  float s = 0.f;
  for (int d = 0; d < 256; ++d) s += r[d] * r[d];
  norms[i] = s;
}

__global__ __launch_bounds__(256) void k_vq_argmin(
    const float* __restrict__ scores, const float* __restrict__ norms,
    const int* __restrict__ cond, int* __restrict__ idx)
{
  int p = blockIdx.x * 256 + threadIdx.x;   // < 32768
  int b = p >> 10;
  const float* nr = norms + (size_t)cond[b] * 512;
  const float* sr = scores + (size_t)p * 512;
  float best = 3.4e38f; int bi = 0;
  for (int k = 0; k < 512; ++k) {
    float d = nr[k] - 2.f * sr[k];
    if (d < best) { best = d; bi = k; }
  }
  idx[p] = bi;
}

__global__ __launch_bounds__(256) void k_vq_gather(
    const float* __restrict__ cb, const int* __restrict__ cond,
    const int* __restrict__ idx, float* __restrict__ zq,
    float* __restrict__ dst_nchw, size_t n)
{
  size_t i = (size_t)blockIdx.x * 256 + threadIdx.x;
  if (i >= n) return;
  int c = (int)(i & 255);
  size_t p = i >> 8;
  int b = (int)(p >> 10);
  int rem = (int)(p & 1023);
  int y = rem >> 5, x = rem & 31;
  float v = cb[(((size_t)cond[b] * 512) + idx[p]) * 256 + c];
  zq[i] = v;
  dst_nchw[(((size_t)b * 256 + c) * 32 + y) * 32 + x] = v;
}

// encoder head: Conv2d(1,256,4,2,1), bias folded away by following BN -> bf16 NHWC
__global__ __launch_bounds__(256) void k_enc_conv1(
    const float* __restrict__ x, const float* __restrict__ w, __bf16* __restrict__ out)
{
  size_t i = (size_t)blockIdx.x * 256 + threadIdx.x;   // < 33554432
  int c = (int)(i & 255);
  size_t p = i >> 8;
  int n = (int)(p >> 12);
  int rem = (int)(p & 4095);
  int oh = rem >> 6, ow = rem & 63;
  const float* xb = x + (size_t)n * 16384;
  float acc = 0.f;
#pragma unroll
  for (int kh = 0; kh < 4; ++kh) {
    int ih = oh * 2 - 1 + kh;
    if (ih < 0 || ih >= 128) continue;
#pragma unroll
    for (int kw = 0; kw < 4; ++kw) {
      int iw = ow * 2 - 1 + kw;
      if (iw < 0 || iw >= 128) continue;
      acc += xb[ih * 128 + iw] * w[c * 16 + kh * 4 + kw];
    }
  }
  out[i] = f2bf(acc);
}

// decoder tail: ConvTranspose2d(256,1,4,2,1) + bias + tanh -> d_out x_tilde
__global__ __launch_bounds__(256) void k_dec_convT2(
    const __bf16* __restrict__ in, const float* __restrict__ w,
    const float* __restrict__ b, float* __restrict__ out)
{
  int i = blockIdx.x * 256 + threadIdx.x;   // < 524288
  int n = i >> 14;
  int rem = i & 16383;
  int oy = rem >> 7, ox = rem & 127;
  int a = oy & 1, bb = ox & 1;
  int y = oy >> 1, xx = ox >> 1;
  float acc = 0.f;
  for (int tky = 0; tky < 2; ++tky) {
    int iy = y + tky - 1 + a;
    if (iy < 0 || iy >= 64) continue;
    int ky = a + 2 * tky;
    for (int tkx = 0; tkx < 2; ++tkx) {
      int ix = xx + tkx - 1 + bb;
      if (ix < 0 || ix >= 64) continue;
      int kx = bb + 2 * tkx;
      const __bf16* pin = in + (((size_t)n * 64 + iy) * 64 + ix) * 256;
      int wbase = (3 - ky) * 4 + (3 - kx);
      float s = 0.f;
      for (int c = 0; c < 256; ++c) s += bf2f(pin[c]) * w[c * 16 + wbase];
      acc += s;
    }
  }
  out[i] = tanhf(acc + b[0]);
}

// ---------------------------------------------------------------------------
// Host orchestration
// ---------------------------------------------------------------------------
extern "C" void kernel_launch(void* const* d_in, const int* in_sizes, int n_in,
                              void* d_out, int out_size, void* d_ws, size_t ws_size,
                              hipStream_t stream)
{
  (void)in_sizes; (void)n_in; (void)out_size; (void)ws_size;

  const float* x        = (const float*)d_in[0];
  const int*   cond     = (const int*)d_in[1];   // values 0..15 (x64 disabled in JAX -> int32)
  const float* enc_w1   = (const float*)d_in[2];
  const float* enc_g1   = (const float*)d_in[4];
  const float* enc_bt1  = (const float*)d_in[5];
  const float* enc_w2   = (const float*)d_in[6];
  const float* enc_b2   = (const float*)d_in[7];
  const float* codebook = (const float*)d_in[24];
  const float* dec_w1   = (const float*)d_in[41];
  const float* dec_g1   = (const float*)d_in[43];
  const float* dec_bt1  = (const float*)d_in[44];
  const float* dec_w2   = (const float*)d_in[45];
  const float* dec_b2   = (const float*)d_in[46];
  // resblock param bases: res1=8, res2=16, dres1=25, dres2=33

  const size_t NS = 8388608;    // 32*32*32*256  small activation elems
  const size_t NB = 33554432;   // 32*64*64*256  big activation elems

  char* ws = (char*)d_ws;
  size_t o = 0;
  auto alloc = [&](size_t bytes) { size_t r = o; o += (bytes + 255) & ~(size_t)255; return r; };
  __bf16* A   = (__bf16*)(ws + alloc(NB * 2));      // big bf16 (enc1 raw/act, dec1 raw/act)
  float*  C1  = (float*) (ws + alloc(NS * 4));
  float*  C2  = (float*) (ws + alloc(NS * 4));      // C2+C3 contiguous -> aliased by scores
  float*  C3  = (float*) (ws + alloc(NS * 4));
  __bf16* D1  = (__bf16*)(ws + alloc(NS * 2));      // small bf16 staging
  __bf16* PW1 = (__bf16*)(ws + alloc(4194304));     // big weight pack region
  __bf16* PW2 = (__bf16*)(ws + alloc(262144));      // 1x1 weight pack region
  __bf16* PCB = (__bf16*)(ws + alloc(4194304));     // packed codebook (16 tables)
  float*  NORMS = (float*)(ws + alloc(32768));      // |code|^2 per (cond, code)
  float*  STATS = (float*)(ws + alloc(2048));       // BN sum/sumsq
  int*    IDX   = (int*)  (ws + alloc(131072));     // argmin indices
  float*  SCORES = C2;                              // 67MB spanning C2+C3

  float* out_xt = (float*)d_out;                    // [32,1,128,128]
  float* out_ze = out_xt + 524288;                  // [32,256,32,32] NCHW
  float* out_zq = out_ze + 8388608;                 // [32,256,32,32] NCHW

  auto mkgeom = [](int IH, int IW, int IC, int OH, int OW, int S, int ntaps,
                   int Ktot, int Nout, int OYS, int OYO, int OXS, int OXO,
                   int OHO, int OWO, int sbf) {
    Geom g; memset(&g, 0, sizeof(g));
    g.IH = IH; g.IW = IW; g.IC = IC; g.OH = OH; g.OW = OW; g.S = S;
    g.ntaps = ntaps; g.Ktot = Ktot; g.Nout = Nout;
    g.OYS = OYS; g.OYO = OYO; g.OXS = OXS; g.OXO = OXO; g.OHO = OHO; g.OWO = OWO;
    g.store_bf16 = sbf;
    return g;
  };

  Geom g_conv2 = mkgeom(64, 64, 256, 32, 32, 2, 16, 4096, 256, 1, 0, 1, 0, 32, 32, 0);
  for (int t = 0; t < 16; ++t) { g_conv2.dy[t] = t / 4 - 1; g_conv2.dx[t] = t % 4 - 1; }
  Geom g_3x3 = mkgeom(32, 32, 256, 32, 32, 1, 9, 2304, 256, 1, 0, 1, 0, 32, 32, 0);
  for (int t = 0; t < 9; ++t) { g_3x3.dy[t] = t / 3 - 1; g_3x3.dx[t] = t % 3 - 1; }
  Geom g_1x1 = mkgeom(32, 32, 256, 32, 32, 1, 1, 256, 256, 1, 0, 1, 0, 32, 32, 0);
  Geom g_vq  = mkgeom(32, 32, 256, 32, 32, 1, 1, 256, 512, 1, 0, 1, 0, 32, 32, 0);

  // grid: M blocks = 32768/128 = 256 ; N blocks = Nout/256
  auto resblock = [&](const float* xin, float* r1, float* r2, float* outbuf, int base) {
    const float* w1  = (const float*)d_in[base + 0];
    const float* g1  = (const float*)d_in[base + 2];
    const float* bt1 = (const float*)d_in[base + 3];
    const float* w2  = (const float*)d_in[base + 4];
    const float* g2  = (const float*)d_in[base + 6];
    const float* bt2 = (const float*)d_in[base + 7];
    k_relu_bf16<<<NS / 256, 256, 0, stream>>>(xin, D1, NS);
    PackW pw1{256, 3, 3, 256, 2304};
    pack_conv_w<<<589824 / 256, 256, 0, stream>>>(w1, PW1, pw1, 589824);
    wmma_gemm<<<dim3(256, 1), 256, 0, stream>>>(D1, PW1, r1, nullptr, nullptr, 0, g_3x3);
    hipMemsetAsync(STATS, 0, 2048, stream);
    k_bn_stats<<<512, 256, 0, stream>>>(r1, nullptr, STATS, 64);
    k_bn_apply<<<NS / 256, 256, 0, stream>>>(r1, nullptr, STATS, 1.f / 32768.f,
                                             g1, bt1, 1, nullptr, nullptr, D1, NS);
    PackW pw2{256, 1, 1, 256, 256};
    pack_conv_w<<<65536 / 256, 256, 0, stream>>>(w2, PW2, pw2, 65536);
    wmma_gemm<<<dim3(256, 1), 256, 0, stream>>>(D1, PW2, r2, nullptr, nullptr, 0, g_1x1);
    hipMemsetAsync(STATS, 0, 2048, stream);
    k_bn_stats<<<512, 256, 0, stream>>>(r2, nullptr, STATS, 64);
    k_bn_apply<<<NS / 256, 256, 0, stream>>>(r2, nullptr, STATS, 1.f / 32768.f,
                                             g2, bt2, 0, xin, outbuf, nullptr, NS);
  };

  // ---- one-time per call: codebook prep ----
  pack_cb<<<2097152 / 256, 256, 0, stream>>>(codebook, PCB);
  k_cb_norms<<<32, 256, 0, stream>>>(codebook, NORMS);

  // ---- encoder ----
  k_enc_conv1<<<NB / 256, 256, 0, stream>>>(x, enc_w1, A);
  hipMemsetAsync(STATS, 0, 2048, stream);
  k_bn_stats<<<2048, 256, 0, stream>>>(nullptr, A, STATS, 64);
  k_bn_apply<<<NB / 256, 256, 0, stream>>>(nullptr, A, STATS, 1.f / 131072.f,
                                           enc_g1, enc_bt1, 1, nullptr, nullptr, A, NB);

  PackW pwc2{256, 4, 4, 256, 4096};
  pack_conv_w<<<1048576 / 256, 256, 0, stream>>>(enc_w2, PW1, pwc2, 1048576);
  wmma_gemm<<<dim3(256, 1), 256, 0, stream>>>(A, PW1, C1, nullptr, nullptr, 0, g_conv2);
  k_add_bias<<<NS / 256, 256, 0, stream>>>(C1, enc_b2, NS);

  resblock(C1, C2, C3, C2, 8);    // -> C2
  resblock(C2, C1, C3, C1, 16);   // z_e -> C1

  // ---- VQ ----
  k_export_small<<<NS / 256, 256, 0, stream>>>(C1, out_ze, D1, NS);
  wmma_gemm<<<dim3(256, 2), 256, 0, stream>>>(D1, PCB, SCORES, nullptr, cond, 131072, g_vq);
  k_vq_argmin<<<128, 256, 0, stream>>>(SCORES, NORMS, cond, IDX);
  k_vq_gather<<<NS / 256, 256, 0, stream>>>(codebook, cond, IDX, C1, out_zq, NS);

  // ---- decoder ----
  resblock(C1, C2, C3, C2, 25);   // -> C2
  resblock(C2, C1, C3, C1, 33);   // -> C1
  k_relu_bf16<<<NS / 256, 256, 0, stream>>>(C1, D1, NS);

  for (int a = 0; a < 2; ++a) {
    for (int b2 = 0; b2 < 2; ++b2) {
      pack_convT<<<262144 / 256, 256, 0, stream>>>(dec_w1, PW1, 256, a, b2, 262144);
      Geom gct = mkgeom(32, 32, 256, 32, 32, 1, 4, 1024, 256, 2, a, 2, b2, 64, 64, 1);
      for (int t = 0; t < 4; ++t) {
        gct.dy[t] = (t >> 1) - 1 + a;
        gct.dx[t] = (t & 1) - 1 + b2;
      }
      wmma_gemm<<<dim3(256, 1), 256, 0, stream>>>(D1, PW1, nullptr, A, nullptr, 0, gct);
    }
  }
  hipMemsetAsync(STATS, 0, 2048, stream);
  k_bn_stats<<<2048, 256, 0, stream>>>(nullptr, A, STATS, 64);
  k_bn_apply<<<NB / 256, 256, 0, stream>>>(nullptr, A, STATS, 1.f / 131072.f,
                                           dec_g1, dec_bt1, 1, nullptr, nullptr, A, NB);
  k_dec_convT2<<<2048, 256, 0, stream>>>(A, dec_w2, dec_b2, out_xt);
}
